// ProteinGAT_30666066493704
// MI455X (gfx1250) — compile-verified, matrix operand order checked by reading, not run on hardware
//
#include <hip/hip_runtime.h>
#include <stdint.h>

#define N_NODES   50000
#define N_EDGES   800000
#define EN        (N_EDGES + N_NODES)
#define D         256
#define D_OUT     323
#define N_GRAPHS  128
#define LN_EPS    1e-5f
#define NEG_SLOPE 0.2f

// LDS layout for k_gemm (all dynamic): [0,16384) = A tile bf16, [16384, 147456) = B (Wt) bf16
#define LDS_A_OFF   0
#define LDS_B_OFF   16384
#define LDS_BYTES   (16384 + 131072)

typedef __bf16 bf16_t;
typedef bf16_t       v16bf __attribute__((ext_vector_type(16)));
typedef float        v8f   __attribute__((ext_vector_type(8)));
typedef unsigned int u32x4 __attribute__((ext_vector_type(4)));
typedef int          i32x4 __attribute__((ext_vector_type(4)));
typedef int          i32x8 __attribute__((ext_vector_type(8)));

#if defined(__has_builtin)
#  if __has_builtin(__builtin_amdgcn_tensor_load_to_lds) && \
      __has_builtin(__builtin_amdgcn_s_wait_tensorcnt)
#    define HAVE_TDM 1
#  else
#    define HAVE_TDM 0
#  endif
#else
#  define HAVE_TDM 0
#endif

static __device__ __forceinline__ unsigned short f2bf(float f) {
  unsigned u = __float_as_uint(f);
  u += 0x7fffu + ((u >> 16) & 1u);          // round-to-nearest-even
  return (unsigned short)(u >> 16);
}
static __device__ __forceinline__ float bf2f(unsigned short b) {
  return __uint_as_float(((unsigned)b) << 16);
}
// monotonic float <-> uint transform for atomicMax-based segment max
static __device__ __forceinline__ unsigned ford(float f) {
  unsigned u = __float_as_uint(f);
  return (u >> 31) ? ~u : (u | 0x80000000u);
}
static __device__ __forceinline__ float ordf(unsigned k) {
  unsigned u = (k >> 31) ? (k & 0x7fffffffu) : ~k;
  return __uint_as_float(u);
}
static __device__ __forceinline__ float leaky(float v) {
  return v > 0.0f ? v : NEG_SLOPE * v;
}

// ---------------------------------------------------------------------------
// W[f32, K x N, row-major] -> Wt[bf16, N x K] so each WMMA B-fragment lane
// reads 16 contiguous bf16 (32B) along K.
__global__ __launch_bounds__(256)
void k_convert_W(const float* __restrict__ W, unsigned short* __restrict__ Wt) {
  int idx = blockIdx.x * 256 + threadIdx.x;     // 65536 total
  int n = idx >> 8, k = idx & 255;
  Wt[idx] = f2bf(W[k * D + n]);
}

// ---------------------------------------------------------------------------
// h = x @ W, bf16 WMMA with f32 accumulate, h stored as bf16.
// Block: 256 threads = 8 waves; tile 32 rows x 256 cols.
// Wave w: rows (w>>2)*16..+15, cols (w&3)*64..+63 (4 accumulators of 16x16).
// B (Wt, 128KB bf16) is staged once per workgroup into LDS by the Tensor Data
// Mover (TENSORcnt); A tile is converted f32->bf16 cooperatively in parallel.
__global__ __launch_bounds__(256)
void k_gemm(const float* __restrict__ x, const unsigned short* __restrict__ Wt,
            unsigned short* __restrict__ h) {
  extern __shared__ __attribute__((aligned(32))) char smem[];
  unsigned short (*At)[D] = (unsigned short (*)[D])(smem + LDS_A_OFF);
  unsigned short* Bs      = (unsigned short*)(smem + LDS_B_OFF);

  const int tid  = threadIdx.x;
  const int row0 = blockIdx.x * 32;

#if HAVE_TDM
  if (tid < 32) {  // wave 0 issues one TDM descriptor: full 256x256 bf16 tile
    unsigned long long ga = (unsigned long long)(const void*)Wt;
    u32x4 g0;
    g0[0] = 1u;                                        // count=1, user D#
    g0[1] = (unsigned)LDS_B_OFF;                       // lds_addr (bytes)
    g0[2] = (unsigned)(ga & 0xFFFFFFFFu);              // global_addr[31:0]
    g0[3] = (unsigned)((ga >> 32) & 0x01FFFFFFu)       // global_addr[56:32]
            | (2u << 30);                              // type=2 ("image")
    i32x8 g1;
    g1[0] = (int)(1u << 16);                           // data_size=1 (2B), mask=0
    g1[1] = (int)((unsigned)D << 16);                  // tensor_dim0[15:0] @63:48
    g1[2] = (int)((unsigned)D << 16);                  // dim0 hi=0 | tensor_dim1 lo
    g1[3] = (int)((unsigned)D << 16);                  // dim1 hi=0 | tile_dim0=256
    g1[4] = D;                                         // tile_dim1=256, tile_dim2=0
    g1[5] = D;                                         // tensor_dim0_stride=256
    g1[6] = 0;
    g1[7] = 0;
    i32x4 z4 = {0, 0, 0, 0};
#if __clang_major__ >= 23
    i32x8 z8 = {0, 0, 0, 0, 0, 0, 0, 0};
    __builtin_amdgcn_tensor_load_to_lds(g0, g1, z4, z4, z8, 0);
#else
    __builtin_amdgcn_tensor_load_to_lds(g0, g1, z4, z4, 0);
#endif
  }
#endif

  // Stage x tile into LDS as bf16 (f32 loads, convert, ds stores) — overlaps TDM
  for (int i = tid; i < 32 * (D / 4); i += 256) {
    int r  = i / (D / 4);
    int c4 = (i % (D / 4)) * 4;
    float4 v = make_float4(0.f, 0.f, 0.f, 0.f);
    if (row0 + r < N_NODES)
      v = *(const float4*)(x + (size_t)(row0 + r) * D + c4);
    ushort4 b;
    b.x = f2bf(v.x); b.y = f2bf(v.y); b.z = f2bf(v.z); b.w = f2bf(v.w);
    *(ushort4*)&At[r][c4] = b;
  }

#if HAVE_TDM
  if (tid < 32) __builtin_amdgcn_s_wait_tensorcnt(0);
#else
  for (int i = tid; i < (D * D) / 8; i += 256)          // cooperative fallback
    ((uint4*)Bs)[i] = ((const uint4*)Wt)[i];
#endif
  __syncthreads();

  const int w    = tid >> 5;
  const int lane = tid & 31;
  const int mt   = (w >> 2) * 16;      // row sub-tile base within block tile
  const int nb   = (w & 3) * 64;       // col base
  const int lr   = lane & 15;
  const int kh   = (lane >> 4) << 4;   // lanes 16..31 hold upper K half

  v8f acc[4] = {v8f{}, v8f{}, v8f{}, v8f{}};

  for (int kt = 0; kt < D; kt += 32) {
    v16bf a = *(const v16bf*)&At[mt + lr][kt + kh];
#pragma unroll
    for (int t = 0; t < 4; ++t) {
      int n = nb + t * 16 + lr;
      v16bf b = *(const v16bf*)(Bs + (size_t)n * D + kt + kh);
      acc[t] = __builtin_amdgcn_wmma_f32_16x16x32_bf16(
          false, a, false, b, (short)0, acc[t], false, false);
    }
  }

#pragma unroll
  for (int t = 0; t < 4; ++t) {
#pragma unroll
    for (int i = 0; i < 8; ++i) {
      int r = row0 + mt + i + ((lane >> 4) << 3);  // VGPR i: M=i (lanes<16) / M=8+i
      int c = nb + t * 16 + lr;
      if (r < N_NODES) h[(size_t)r * D + c] = f2bf(acc[t][i]);
    }
  }
}

// ---------------------------------------------------------------------------
// a_src[r] = h[r,:].att_src ; a_dst[r] = h[r,:].att_dst  (one wave per row)
__global__ __launch_bounds__(256)
void k_att(const unsigned short* __restrict__ h, const float* __restrict__ att_src,
           const float* __restrict__ att_dst, float* __restrict__ a_src,
           float* __restrict__ a_dst) {
  int r    = (int)(((size_t)blockIdx.x * 256 + threadIdx.x) >> 5);
  int lane = threadIdx.x & 31;
  if (r >= N_NODES) return;
  unsigned short hl[8] __attribute__((aligned(16)));
  *(uint4*)hl = *(const uint4*)(h + (size_t)r * D + lane * 8);
  float s0 = 0.f, s1 = 0.f;
#pragma unroll
  for (int i = 0; i < 8; ++i) {
    float v = bf2f(hl[i]);
    s0 += v * att_src[lane * 8 + i];
    s1 += v * att_dst[lane * 8 + i];
  }
  for (int m = 16; m; m >>= 1) {
    s0 += __shfl_xor(s0, m, 32);
    s1 += __shfl_xor(s1, m, 32);
  }
  if (lane == 0) { a_src[r] = s0; a_dst[r] = s1; }
}

// ---------------------------------------------------------------------------
__global__ __launch_bounds__(256)
void k_init_nodes(unsigned* __restrict__ emax_u, float* __restrict__ denom) {
  int i = blockIdx.x * 256 + threadIdx.x;
  if (i < N_NODES) { emax_u[i] = 0u; denom[i] = 0.f; }
}

__global__ __launch_bounds__(256)
void k_init_agg(const float* __restrict__ x, const float* __restrict__ bias,
                float* __restrict__ agg) {
  size_t i = (size_t)blockIdx.x * 256 + threadIdx.x;   // N*D threads exactly
  agg[i] = x[i] + bias[i & (D - 1)];
}

__global__ __launch_bounds__(256)
void k_init_pool(float* __restrict__ psum, float* __restrict__ cnt) {
  int i = blockIdx.x * 256 + threadIdx.x;
  if (i < N_GRAPHS * D) psum[i] = 0.f;
  if (i < N_GRAPHS) cnt[i] = 0.f;
}

// ---------------------------------------------------------------------------
static __device__ __forceinline__ void edge_sd(const int* ei, int e, int& s, int& d) {
  if (e < N_EDGES) { s = ei[e]; d = ei[N_EDGES + e]; }
  else             { s = d = e - N_EDGES; }            // self loops
}

__global__ __launch_bounds__(256)
void k_edge_max(const int* __restrict__ ei, const float* __restrict__ a_src,
                const float* __restrict__ a_dst, unsigned* __restrict__ emax_u) {
  int e = blockIdx.x * 256 + threadIdx.x;
  if (e >= EN) return;
  int s, d; edge_sd(ei, e, s, d);
  atomicMax(emax_u + d, ford(leaky(a_src[s] + a_dst[d])));
}

__global__ __launch_bounds__(256)
void k_edge_sum(const int* __restrict__ ei, const float* __restrict__ a_src,
                const float* __restrict__ a_dst, const unsigned* __restrict__ emax_u,
                float* __restrict__ denom) {
  int e = blockIdx.x * 256 + threadIdx.x;
  if (e >= EN) return;
  int s, d; edge_sd(ei, e, s, d);
  float ee = __expf(leaky(a_src[s] + a_dst[d]) - ordf(emax_u[d]));
  atomicAdd(denom + d, ee);
}

// One wave per edge: gather bf16 h[src] row (16B/lane), scatter f32 atomics.
__global__ __launch_bounds__(256)
void k_edge_agg(const int* __restrict__ ei, const float* __restrict__ a_src,
                const float* __restrict__ a_dst, const unsigned* __restrict__ emax_u,
                const float* __restrict__ denom, const unsigned short* __restrict__ h,
                float* __restrict__ agg) {
  int e    = (int)(((size_t)blockIdx.x * 256 + threadIdx.x) >> 5);
  int lane = threadIdx.x & 31;
  if (e >= EN) return;
  int s, d; edge_sd(ei, e, s, d);
  float t = leaky(a_src[s] + a_dst[d]);
  float alpha = __expf(t - ordf(emax_u[d])) / denom[d];
  unsigned short hl[8] __attribute__((aligned(16)));
  *(uint4*)hl = *(const uint4*)(h + (size_t)s * D + lane * 8);
  float* ap = agg + (size_t)d * D + lane * 8;
#pragma unroll
  for (int i = 0; i < 8; ++i) atomicAdd(ap + i, alpha * bf2f(hl[i]));
}

// ---------------------------------------------------------------------------
// LayerNorm(z) fused with per-graph sum pooling. One wave per row.
__global__ __launch_bounds__(256)
void k_ln_pool(const float* __restrict__ agg, const int* __restrict__ batch,
               const float* __restrict__ gamma, const float* __restrict__ beta,
               float* __restrict__ psum, float* __restrict__ cnt) {
  int r    = (int)(((size_t)blockIdx.x * 256 + threadIdx.x) >> 5);
  int lane = threadIdx.x & 31;
  if (r >= N_NODES) return;
  const float* zr = agg + (size_t)r * D + lane * 8;
  float v[8] __attribute__((aligned(16)));
  *(float4*)v       = *(const float4*)zr;
  *(float4*)(v + 4) = *(const float4*)(zr + 4);
  float s = 0.f, ss = 0.f;
#pragma unroll
  for (int i = 0; i < 8; ++i) { s += v[i]; ss += v[i] * v[i]; }
  for (int m = 16; m; m >>= 1) {
    s  += __shfl_xor(s, m, 32);
    ss += __shfl_xor(ss, m, 32);
  }
  float mu  = s * (1.f / D);
  float var = ss * (1.f / D) - mu * mu;
  float rs  = rsqrtf(var + LN_EPS);
  int g = batch[r];
  float* pp = psum + (size_t)g * D + lane * 8;
#pragma unroll
  for (int i = 0; i < 8; ++i) {
    int c = lane * 8 + i;
    float zi = (v[i] - mu) * rs * gamma[c] + beta[c];
    atomicAdd(pp + i, zi);
  }
  if (lane == 0) atomicAdd(cnt + g, 1.0f);
}

// ---------------------------------------------------------------------------
// out[g,o] = (psum[g,:]/max(cnt,1)) . cls_W[:,o] + cls_b[o]
__global__ __launch_bounds__(256)
void k_cls(const float* __restrict__ psum, const float* __restrict__ cnt,
           const float* __restrict__ cls_W, const float* __restrict__ cls_b,
           float* __restrict__ out) {
  __shared__ float p[D];
  int g = blockIdx.y;
  int o = blockIdx.x * 256 + threadIdx.x;
  float inv = 1.0f / fmaxf(cnt[g], 1.0f);
  p[threadIdx.x] = psum[(size_t)g * D + threadIdx.x] * inv;
  __syncthreads();
  if (o < D_OUT) {
    float acc = cls_b[o];
#pragma unroll 8
    for (int d = 0; d < D; ++d) acc += p[d] * cls_W[(size_t)d * D_OUT + o];
    out[(size_t)g * D_OUT + o] = acc;
  }
}

// ---------------------------------------------------------------------------
extern "C" void kernel_launch(void* const* d_in, const int* in_sizes, int n_in,
                              void* d_out, int out_size, void* d_ws, size_t ws_size,
                              hipStream_t stream) {
  const float* x       = (const float*)d_in[0];
  const int*   ei      = (const int*)d_in[1];
  const int*   batch   = (const int*)d_in[2];
  const float* W       = (const float*)d_in[3];
  const float* att_src = (const float*)d_in[4];
  const float* att_dst = (const float*)d_in[5];
  const float* bias    = (const float*)d_in[6];
  const float* gamma   = (const float*)d_in[7];
  const float* beta    = (const float*)d_in[8];
  const float* cls_W   = (const float*)d_in[9];
  const float* cls_b   = (const float*)d_in[10];
  float* out = (float*)d_out;

  char* ws = (char*)d_ws;
  size_t off = 0;
  auto alloc = [&](size_t bytes) -> void* {
    void* p = ws + off;
    off = (off + bytes + 255) & ~(size_t)255;
    return p;
  };
  unsigned short* h      = (unsigned short*)alloc((size_t)N_NODES * D * 2);
  unsigned short* Wt     = (unsigned short*)alloc((size_t)D * D * 2);
  float*          a_src  = (float*)alloc((size_t)N_NODES * 4);
  float*          a_dst  = (float*)alloc((size_t)N_NODES * 4);
  unsigned*       emax_u = (unsigned*)alloc((size_t)N_NODES * 4);
  float*          denom  = (float*)alloc((size_t)N_NODES * 4);
  float*          agg    = (float*)alloc((size_t)N_NODES * D * 4);
  float*          psum   = (float*)alloc((size_t)N_GRAPHS * D * 4);
  float*          cnt    = (float*)alloc((size_t)N_GRAPHS * 4);
  (void)ws_size; (void)in_sizes; (void)n_in; (void)out_size;

  const int B = 256;
  k_convert_W<<<(D * D) / B, B, 0, stream>>>(W, Wt);
  k_gemm<<<(N_NODES + 31) / 32, B, LDS_BYTES, stream>>>(x, Wt, h);
  k_att<<<((size_t)N_NODES * 32 + B - 1) / B, B, 0, stream>>>(h, att_src, att_dst, a_src, a_dst);
  k_init_nodes<<<(N_NODES + B - 1) / B, B, 0, stream>>>(emax_u, denom);
  k_init_agg<<<((size_t)N_NODES * D) / B, B, 0, stream>>>(x, bias, agg);
  k_init_pool<<<(N_GRAPHS * D + B - 1) / B, B, 0, stream>>>(psum, cnt);
  k_edge_max<<<(EN + B - 1) / B, B, 0, stream>>>(ei, a_src, a_dst, emax_u);
  k_edge_sum<<<(EN + B - 1) / B, B, 0, stream>>>(ei, a_src, a_dst, emax_u, denom);
  k_edge_agg<<<((size_t)EN * 32 + B - 1) / B, B, 0, stream>>>(ei, a_src, a_dst, emax_u,
                                                             denom, h, agg);
  k_ln_pool<<<((size_t)N_NODES * 32 + B - 1) / B, B, 0, stream>>>(agg, batch, gamma, beta,
                                                                  psum, cnt);
  dim3 gcls((D_OUT + B - 1) / B, N_GRAPHS);
  k_cls<<<gcls, B, 0, stream>>>(psum, cnt, cls_W, cls_b, out);
}
